// LOUPEMask_21311627723003
// MI455X (gfx1250) — compile-verified, matrix operand order checked by reading, not run on hardware
//
#include <hip/hip_runtime.h>

// ---------------------------------------------------------------------------
// LOUPE mask: pmask = rescale(sigmoid(5*w)); per-row top-K(64) of pmask-thresh
// builds a binary mask [B,S]; output = example * mask broadcast over (C,H).
//
// Stage 1 (tiny): top-K via matrix-pipe rank reduction (v_wmma_f32_16x16x32_f16
//                 with an all-ones A and a 0/1 comparison B; exact counts).
// Stage 2 (hot):  1.07 GB pure streaming, nontemporal b128 load/mul/store.
// ---------------------------------------------------------------------------

typedef __attribute__((ext_vector_type(16))) _Float16 v16h;
typedef __attribute__((ext_vector_type(8)))  float    v8f;
typedef __attribute__((ext_vector_type(4)))  float    f32x4;

#define S_DIM   512
#define B_DIM   64
#define CH_DIM  4096          // C*H = 8*512
#define K_TOP   64
#define SLOPE   5.0f
#define SPARS   0.125f

// ---------------------------------------------------------------------------
// Kernel 1: one block per batch row b (512 threads = 16 waves).
//  - pmask recomputed per block (512 sigmoids + LDS mean: negligible)
//  - score[s] into LDS
//  - rank[s] = #{ j : score[j] > score[s] || (== && j < s) } computed on the
//    matrix pipe: acc(16x16 f32) += ones(16x32 f16) x Cmp(32x16 f16),
//    16 K-chunks cover all 512 j. All-ones A => sum is invariant to the
//    K-slot ordering inside the B operand; only lane->N matters.
//  - mask[b,s] = (rank < 64)
// ---------------------------------------------------------------------------
__global__ __launch_bounds__(512)
void loupe_mask_topk(const float* __restrict__ weight,
                     const float* __restrict__ thresh,
                     float* __restrict__ mask)
{
    __shared__ float red[S_DIM];
    __shared__ float score[S_DIM];

    const int tid = threadIdx.x;           // 0..511, == s for stage A
    const int b   = blockIdx.x;            // 0..63

    // sigmoid(5*w)
    const float sig = 1.0f / (1.0f + __expf(-SLOPE * weight[tid]));

    // mean over S via LDS tree
    red[tid] = sig;
    __syncthreads();
    for (int off = 256; off > 0; off >>= 1) {
        if (tid < off) red[tid] += red[tid + off];
        __syncthreads();
    }
    const float xbar = red[0] * (1.0f / (float)S_DIM);

    // rescale_prob
    const float hi = sig * (SPARS / xbar);
    const float lo = 1.0f - (1.0f - sig) * ((1.0f - SPARS) / (1.0f - xbar));
    const float pm = (xbar > SPARS) ? hi : lo;

    score[tid] = pm - thresh[b * S_DIM + tid];
    __syncthreads();

    const int lane  = tid & 31;            // wave32
    const int wave  = tid >> 5;            // 0..15
    const int ncol  = lane & 15;           // output column within tile
    const int khalf = (lane >> 4) << 4;    // lanes L / L+16 split 32 j's

    v16h ones;
#pragma unroll
    for (int t = 0; t < 16; ++t) ones[t] = (_Float16)1.0f;

    // 16 waves x 2 tiles of 16 columns = 512 s-values
    for (int tset = 0; tset < 2; ++tset) {
        const int   tile = wave + tset * 16;
        const int   scol = tile * 16 + ncol;
        const float ss   = score[scol];

        v8f acc = {};
        for (int kc = 0; kc < 16; ++kc) {      // 16 chunks x 32 j = 512
            const int jb = kc * 32 + khalf;
            v16h bm;
#pragma unroll
            for (int t = 0; t < 16; ++t) {
                const int   j  = jb + t;
                const float sj = score[j];
                const bool  gt = (sj > ss) || ((sj == ss) && (j < scol));
                bm[t] = gt ? (_Float16)1.0f : (_Float16)0.0f;
            }
            // D = ones(16x32) * Cmp(32x16) + acc : every row of D = rank[n]
            acc = __builtin_amdgcn_wmma_f32_16x16x32_f16(
                      false, ones, false, bm, (short)0, acc, false, false);
        }

        const float rank = acc[0];             // exact integer count in f32
        if (lane < 16)
            mask[b * S_DIM + scol] = (rank < (float)K_TOP) ? 1.0f : 0.0f;
    }
}

// ---------------------------------------------------------------------------
// Kernel 2: bandwidth-bound mask application.
//   grid = (64 b, 128 chunks), 256 threads, 16 iters x float4 per thread.
//   Thread's s-phase is iteration-invariant -> mask float4 hoisted (one
//   cached/L2 load), hot loop is NT b128 load -> mul -> NT b128 store.
// ---------------------------------------------------------------------------
__global__ __launch_bounds__(256)
void loupe_apply(const float* __restrict__ ex,
                 const float* __restrict__ mask,
                 float* __restrict__ out)
{
    const int b     = blockIdx.x;          // 0..63
    const int chunk = blockIdx.y;          // 0..127
    const int t     = threadIdx.x;         // 0..255

    const size_t base = (size_t)b * ((size_t)CH_DIM * S_DIM)
                      + (size_t)chunk * 16384;      // 16384 = 16*256*4 elems

    // s of this thread's float4 is the same every iteration (1024 % 512 == 0)
    const int   s4 = (t * 4) & (S_DIM - 1);
    const f32x4 m  = *(const f32x4*)(mask + b * S_DIM + s4);  // L2-resident

#pragma unroll
    for (int it = 0; it < 16; ++it) {
        const size_t off = base + (size_t)it * 1024 + (size_t)t * 4;
        f32x4 x = __builtin_nontemporal_load((const f32x4*)(ex + off));
        f32x4 r = x * m;
        __builtin_nontemporal_store(r, (f32x4*)(out + off));
    }
}

// ---------------------------------------------------------------------------
extern "C" void kernel_launch(void* const* d_in, const int* in_sizes, int n_in,
                              void* d_out, int out_size, void* d_ws, size_t ws_size,
                              hipStream_t stream)
{
    const float* ex = (const float*)d_in[0];   // [64, 8, 512, 512] f32
    const float* w  = (const float*)d_in[1];   // [512] f32
    const float* th = (const float*)d_in[2];   // [64, 512] f32
    float* out  = (float*)d_out;               // [64, 8, 512, 512] f32
    float* mask = (float*)d_ws;                // [64, 512] f32 scratch (128 KB)

    (void)in_sizes; (void)n_in; (void)out_size; (void)ws_size;

    loupe_mask_topk<<<dim3(B_DIM), dim3(512), 0, stream>>>(w, th, mask);
    loupe_apply<<<dim3(B_DIM, 128), dim3(256), 0, stream>>>(ex, mask, out);
}
